// RefactorNeighborhoodAttention2D_43954695307309
// MI455X (gfx1250) — compile-verified, matrix operand order checked by reading, not run on hardware
//
#include <hip/hip_runtime.h>
#include <hip/hip_bf16.h>

// ---------------------------------------------------------------------------
// Types
// ---------------------------------------------------------------------------
typedef __attribute__((ext_vector_type(16))) __bf16 v16bf;
typedef __attribute__((ext_vector_type(8)))  __bf16 v8bf;
typedef __attribute__((ext_vector_type(8)))  float  v8f;
typedef unsigned int __attribute__((ext_vector_type(4))) u32x4;
typedef int          __attribute__((ext_vector_type(4))) i32x4;
typedef int          __attribute__((ext_vector_type(8))) i32x8;

#define NB   4
#define HH_  128
#define WW_  128
#define EE_  256
#define NH_  8
#define HD_  32
#define WS_  7
#define PAD_ 3
#define HW_  (HH_*WW_)          // 16384
#define NTOK (NB*HW_)           // 65536
#define NE_  ((size_t)NTOK*EE_) // 16777216 elements

// ---------------------------------------------------------------------------
// bf16 helpers (RNE)
// ---------------------------------------------------------------------------
__device__ __forceinline__ unsigned int bfbits(float f) {
    unsigned int u = __float_as_uint(f);
    return ((u + 0x7FFFu + ((u >> 16) & 1u)) >> 16) & 0xFFFFu;
}
__device__ __forceinline__ __bf16 f2bf(float f) {
    unsigned short h = (unsigned short)bfbits(f);
    union { unsigned short s; __bf16 b; } c; c.s = h; return c.b;
}
__device__ __forceinline__ float lo_bf(unsigned int u) { return __uint_as_float(u << 16); }
__device__ __forceinline__ float hi_bf(unsigned int u) { return __uint_as_float(u & 0xFFFF0000u); }

// ---------------------------------------------------------------------------
// fp32 -> bf16 conversion (n multiple of 1024)
// ---------------------------------------------------------------------------
__global__ __launch_bounds__(256) void cvt_bf16(const float* __restrict__ s,
                                                __bf16* __restrict__ d, int n) {
    int i = (blockIdx.x * 256 + threadIdx.x) * 4;
    if (i >= n) return;
    float4 f = *(const float4*)(s + i);
    unsigned int u0 = bfbits(f.x) | (bfbits(f.y) << 16);
    unsigned int u1 = bfbits(f.z) | (bfbits(f.w) << 16);
    uint2 uu; uu.x = u0; uu.y = u1;
    *(uint2*)(d + i) = uu;
}

// ---------------------------------------------------------------------------
// WMMA fragment loader (16-bit A 16x32 lane map, ISA 7.12.2).
// lane l16 = row; hi selects K-chunks {hi*8..+7} and {16+hi*8..+7}.
// Row-major W[n][k] as B uses identical addressing with row = n.
// ---------------------------------------------------------------------------
__device__ __forceinline__ v16bf load_frag(const __bf16* __restrict__ base,
                                           int row, int kb, int hi) {
    const __bf16* p = base + (size_t)row * EE_ + kb + hi * 8;
    v8bf lo = *(const v8bf*)p;          // 16B aligned
    v8bf h8 = *(const v8bf*)(p + 16);
    v16bf f;
#pragma unroll
    for (int i = 0; i < 8; ++i) { f[i] = lo[i]; f[i + 8] = h8[i]; }
    return f;
}

// ---------------------------------------------------------------------------
// Fused Q|K|V projection GEMM:  y = x @ W^T + b   (M=65536, N=768, K=256)
// grid (2048, 3), block 128 (4 waves). blockIdx.y = matrix select (Q/K/V).
// Each wave register-blocks a 32x64 tile: 2 A-frags x 4 B-frags -> 8 WMMA
// per K-step from 12 b128 loads (1.5 loads/WMMA).
// Q is pre-scaled by 1/sqrt(HD); output scattered to [b][head][h*w][hd].
// ---------------------------------------------------------------------------
__global__ __launch_bounds__(128) void qkv_gemm(
        const __bf16* __restrict__ xb,
        const __bf16* __restrict__ wq, const __bf16* __restrict__ wk,
        const __bf16* __restrict__ wv,
        const float* __restrict__ bq, const float* __restrict__ bk,
        const float* __restrict__ bv,
        __bf16* __restrict__ q, __bf16* __restrict__ k, __bf16* __restrict__ v) {
    const int lane  = threadIdx.x & 31;
    const int wid   = threadIdx.x >> 5;       // 0..3
    const int l16   = lane & 15;
    const int hi    = lane >> 4;
    const int mbase = blockIdx.x * 32;
    const int n0    = wid * 64;               // 0..192 within one matrix
    const int sel   = blockIdx.y;             // 0=Q 1=K 2=V

    const __bf16* W    = (sel == 0) ? wq : ((sel == 1) ? wk : wv);
    const float*  bias = (sel == 0) ? bq : ((sel == 1) ? bk : bv);
    __bf16*       dst  = (sel == 0) ? q  : ((sel == 1) ? k  : v);

    v8f acc[2][4] = {};
#pragma unroll
    for (int kb = 0; kb < EE_; kb += 32) {
        v16bf a[2], b[4];
#pragma unroll
        for (int i = 0; i < 2; ++i) a[i] = load_frag(xb, mbase + i * 16 + l16, kb, hi);
#pragma unroll
        for (int j = 0; j < 4; ++j) b[j] = load_frag(W, n0 + j * 16 + l16, kb, hi);
#pragma unroll
        for (int i = 0; i < 2; ++i)
#pragma unroll
            for (int j = 0; j < 4; ++j)
                acc[i][j] = __builtin_amdgcn_wmma_f32_16x16x32_bf16(
                                false, a[i], false, b[j], (short)0, acc[i][j],
                                false, false);
    }

    const float scale = 0.17677669529663687f;   // 1/sqrt(32)
#pragma unroll
    for (int j = 0; j < 4; ++j) {
        const int   n    = n0 + j * 16 + l16;
        const float bval = bias[n];
        const int   head = n >> 5, d = n & 31;
#pragma unroll
        for (int i = 0; i < 2; ++i)
#pragma unroll
            for (int r = 0; r < 8; ++r) {
                int   row = mbase + i * 16 + r + hi * 8;   // token id
                float val = acc[i][j][r] + bval;
                if (sel == 0) val *= scale;
                int b_  = row >> 14;                       // / HW
                int pix = row & (HW_ - 1);
                dst[((size_t)(b_ * NH_ + head) * HW_ + pix) * HD_ + d] = f2bf(val);
            }
    }
}

// ---------------------------------------------------------------------------
// Manual (VGPR-staged) halo load for border tiles
// ---------------------------------------------------------------------------
#define HALO 22
__device__ __forceinline__ void stage_halo_manual(
        const __bf16* __restrict__ k, const __bf16* __restrict__ v,
        __bf16* kS, __bf16* vS, size_t basebh, int ti0, int tj0, int tid) {
    for (int idx = tid; idx < HALO * HALO; idx += 256) {
        int hy = idx / HALO, hx = idx - hy * HALO;
        int gi = ti0 - PAD_ + hy, gj = tj0 - PAD_ + hx;
        uint4 z; z.x = z.y = z.z = z.w = 0u;
        uint4 a0 = z, a1 = z, a2 = z, a3 = z, b0 = z, b1 = z, b2 = z, b3 = z;
        if (gi >= 0 && gi < HH_ && gj >= 0 && gj < WW_) {
            const uint4* ks = (const uint4*)(k + (basebh + gi * WW_ + gj) * HD_);
            const uint4* vs = (const uint4*)(v + (basebh + gi * WW_ + gj) * HD_);
            a0 = ks[0]; a1 = ks[1]; a2 = ks[2]; a3 = ks[3];
            b0 = vs[0]; b1 = vs[1]; b2 = vs[2]; b3 = vs[3];
        }
        uint4* kd = (uint4*)(kS + (size_t)idx * HD_);
        uint4* vd = (uint4*)(vS + (size_t)idx * HD_);
        kd[0] = a0; kd[1] = a1; kd[2] = a2; kd[3] = a3;
        vd[0] = b0; vd[1] = b1; vd[2] = b2; vd[3] = b3;
    }
}

// ---------------------------------------------------------------------------
// Neighborhood attention (7x7, zero-padded like the reference).
// grid (64 tiles, 32 b*head), block 256: one thread per pixel of a 16x16
// tile; K/V halos (22x22x32 bf16, ~30.3KB each) staged in LDS.
// Interior tiles use the Tensor Data Mover (2D tile: dim0=704 elem rows,
// 22 rows, row stride 4096 elems -> dense in LDS); border tiles (which need
// zero padding) use the manual path.
// ---------------------------------------------------------------------------
__global__ __launch_bounds__(256) void natten(
        const __bf16* __restrict__ q, const __bf16* __restrict__ k,
        const __bf16* __restrict__ v, __bf16* __restrict__ attn) {
    __shared__ __bf16 kS[HALO * HALO * HD_];
    __shared__ __bf16 vS[HALO * HALO * HD_];

    const int bh   = blockIdx.y;               // b*NH + head
    const int tile = blockIdx.x;
    const int ti0  = (tile >> 3) * 16;
    const int tj0  = (tile & 7) * 16;
    const int tid  = threadIdx.x;
    const size_t basebh = (size_t)bh * HW_;

#if __has_builtin(__builtin_amdgcn_tensor_load_to_lds)
    const bool interior = (ti0 >= 16 && ti0 <= 96 && tj0 >= 16 && tj0 <= 96);
    if (interior) {
        if (tid < 32) {   // one wave issues the TDM ops (EXEC ignored by TDM)
            const int gi0 = ti0 - PAD_, gj0 = tj0 - PAD_;
            unsigned long long gaK =
                (unsigned long long)(const char*)(k + (basebh + gi0 * WW_ + gj0) * HD_);
            unsigned long long gaV =
                (unsigned long long)(const char*)(v + (basebh + gi0 * WW_ + gj0) * HD_);
            unsigned ldsK = (unsigned)(uintptr_t)(void*)kS;
            unsigned ldsV = (unsigned)(uintptr_t)(void*)vS;
            // D# group1 (ISA 8.4): data_size=2B; tensor_dim0=2^30, dim1=2^20;
            // tile_dim0=704, tile_dim1=22; tensor_dim0_stride=4096.
            i32x8 g1;
            g1[0] = 0x00010000;                  // mask=0, data_size=1 (2B)
            g1[1] = 0;                           // barrier addr 0 | dim0[15:0]=0
            g1[2] = 0x00004000;                  // dim0[31:16]=0x4000 | dim1[15:0]=0
            g1[3] = (int)(0x10u | (704u << 16)); // dim1[31:16]=0x10 | tile_dim0=704
            g1[4] = 22;                          // tile_dim1=22 | tile_dim2=0
            g1[5] = 4096;                        // tensor_dim0_stride
            g1[6] = 0;
            g1[7] = 0;
            i32x4 gz4 = {0, 0, 0, 0};
            i32x8 gz8 = {0, 0, 0, 0, 0, 0, 0, 0};
            // D# group0 (ISA 8.3): count=1 | lds_addr | global_addr | type=2
            u32x4 g0K = { 1u, ldsK, (unsigned)(gaK & 0xFFFFFFFFu),
                          (unsigned)((gaK >> 32) & 0x01FFFFFFu) | 0x80000000u };
            u32x4 g0V = { 1u, ldsV, (unsigned)(gaV & 0xFFFFFFFFu),
                          (unsigned)((gaV >> 32) & 0x01FFFFFFu) | 0x80000000u };
            __builtin_amdgcn_tensor_load_to_lds(g0K, g1, gz4, gz4, gz8, 0);
            __builtin_amdgcn_tensor_load_to_lds(g0V, g1, gz4, gz4, gz8, 0);
            __builtin_amdgcn_s_wait_tensorcnt(0);
        }
    } else {
        stage_halo_manual(k, v, kS, vS, basebh, ti0, tj0, tid);
    }
#else
    stage_halo_manual(k, v, kS, vS, basebh, ti0, tj0, tid);
#endif
    __syncthreads();

    const int ti = tid >> 4, tj = tid & 15;
    const int gi = ti0 + ti, gj = tj0 + tj;

    // ---- load q (already scaled by 1/sqrt(hd)) ----
    float qr[32];
    {
        const uint4* qp = (const uint4*)(q + (basebh + gi * WW_ + gj) * HD_);
#pragma unroll
        for (int c = 0; c < 4; ++c) {
            uint4 u4 = qp[c];
            unsigned int uu[4] = {u4.x, u4.y, u4.z, u4.w};
#pragma unroll
            for (int t = 0; t < 4; ++t) {
                qr[c * 8 + t * 2]     = lo_bf(uu[t]);
                qr[c * 8 + t * 2 + 1] = hi_bf(uu[t]);
            }
        }
    }

    // ---- scores ----
    float s[49];
#pragma unroll
    for (int dy = 0; dy < WS_; ++dy)
#pragma unroll
        for (int dx = 0; dx < WS_; ++dx) {
            const uint4* kp = (const uint4*)(kS + ((ti + dy) * HALO + (tj + dx)) * HD_);
            float acc = 0.f;
#pragma unroll
            for (int c = 0; c < 4; ++c) {
                uint4 u4 = kp[c];
                unsigned int uu[4] = {u4.x, u4.y, u4.z, u4.w};
#pragma unroll
                for (int t = 0; t < 4; ++t) {
                    int d0 = c * 8 + t * 2;
                    acc = fmaf(qr[d0],     lo_bf(uu[t]), acc);
                    acc = fmaf(qr[d0 + 1], hi_bf(uu[t]), acc);
                }
            }
            s[dy * WS_ + dx] = acc;
        }

    // ---- softmax over all 49 (zero-padded entries included, like ref) ----
    float m = s[0];
#pragma unroll
    for (int o = 1; o < 49; ++o) m = fmaxf(m, s[o]);
    float sum = 0.f;
#pragma unroll
    for (int o = 0; o < 49; ++o) { float e = __expf(s[o] - m); s[o] = e; sum += e; }
    const float inv = 1.0f / sum;

    // ---- weighted V aggregation ----
    float outr[32];
#pragma unroll
    for (int d = 0; d < 32; ++d) outr[d] = 0.f;
#pragma unroll
    for (int dy = 0; dy < WS_; ++dy)
#pragma unroll
        for (int dx = 0; dx < WS_; ++dx) {
            const float w = s[dy * WS_ + dx];
            const uint4* vp = (const uint4*)(vS + ((ti + dy) * HALO + (tj + dx)) * HD_);
#pragma unroll
            for (int c = 0; c < 4; ++c) {
                uint4 u4 = vp[c];
                unsigned int uu[4] = {u4.x, u4.y, u4.z, u4.w};
#pragma unroll
                for (int t = 0; t < 4; ++t) {
                    int d0 = c * 8 + t * 2;
                    outr[d0]     = fmaf(w, lo_bf(uu[t]), outr[d0]);
                    outr[d0 + 1] = fmaf(w, hi_bf(uu[t]), outr[d0 + 1]);
                }
            }
        }

    // ---- store to [token][E] bf16 for the output GEMM ----
    const int b_ = bh >> 3, head = bh & 7;
    const size_t token = (size_t)b_ * HW_ + gi * WW_ + gj;
    unsigned int* op = (unsigned int*)(attn + token * EE_ + head * HD_);
#pragma unroll
    for (int dd = 0; dd < 16; ++dd) {
        unsigned int a = bfbits(outr[2 * dd] * inv);
        unsigned int b = bfbits(outr[2 * dd + 1] * inv);
        op[dd] = a | (b << 16);
    }
}

// ---------------------------------------------------------------------------
// Output projection GEMM:  out = attn @ Wo^T + bo  (fp32 out)
// grid (2048), block 128 (4 waves x 64 cols = 256 cols). 32x64 per wave.
// ---------------------------------------------------------------------------
__global__ __launch_bounds__(128) void out_gemm(
        const __bf16* __restrict__ attn, const __bf16* __restrict__ wo,
        const float* __restrict__ bo, float* __restrict__ out) {
    const int lane  = threadIdx.x & 31;
    const int wid   = threadIdx.x >> 5;
    const int l16   = lane & 15;
    const int hi    = lane >> 4;
    const int mbase = blockIdx.x * 32;
    const int n0    = wid * 64;

    v8f acc[2][4] = {};
#pragma unroll
    for (int kb = 0; kb < EE_; kb += 32) {
        v16bf a[2], b[4];
#pragma unroll
        for (int i = 0; i < 2; ++i) a[i] = load_frag(attn, mbase + i * 16 + l16, kb, hi);
#pragma unroll
        for (int j = 0; j < 4; ++j) b[j] = load_frag(wo, n0 + j * 16 + l16, kb, hi);
#pragma unroll
        for (int i = 0; i < 2; ++i)
#pragma unroll
            for (int j = 0; j < 4; ++j)
                acc[i][j] = __builtin_amdgcn_wmma_f32_16x16x32_bf16(
                                false, a[i], false, b[j], (short)0, acc[i][j],
                                false, false);
    }

#pragma unroll
    for (int j = 0; j < 4; ++j) {
        const int   n    = n0 + j * 16 + l16;
        const float bval = bo[n];
#pragma unroll
        for (int i = 0; i < 2; ++i)
#pragma unroll
            for (int r = 0; r < 8; ++r) {
                int row = mbase + i * 16 + r + hi * 8;
                out[(size_t)row * EE_ + n] = acc[i][j][r] + bval;
            }
    }
}

// ---------------------------------------------------------------------------
// Launcher
// ---------------------------------------------------------------------------
extern "C" void kernel_launch(void* const* d_in, const int* in_sizes, int n_in,
                              void* d_out, int out_size, void* d_ws, size_t ws_size,
                              hipStream_t stream) {
    const float* x  = (const float*)d_in[0];
    const float* Wq = (const float*)d_in[1];
    const float* bq = (const float*)d_in[2];
    const float* Wk = (const float*)d_in[3];
    const float* bk = (const float*)d_in[4];
    const float* Wv = (const float*)d_in[5];
    const float* bv = (const float*)d_in[6];
    const float* Wo = (const float*)d_in[7];
    const float* bo = (const float*)d_in[8];

    const size_t EEe = (size_t)EE_ * EE_;   // 65536 weight elems
    char* w = (char*)d_ws;
    __bf16* xb  = (__bf16*)w; w += NE_ * 2;
    __bf16* wqb = (__bf16*)w; w += EEe * 2;
    __bf16* wkb = (__bf16*)w; w += EEe * 2;
    __bf16* wvb = (__bf16*)w; w += EEe * 2;
    __bf16* wob = (__bf16*)w; w += EEe * 2;
    __bf16* qb  = (__bf16*)w; w += NE_ * 2;
    __bf16* kb_ = (__bf16*)w; w += NE_ * 2;
    __bf16* vb_ = (__bf16*)w; w += NE_ * 2;
    __bf16* ab  = (__bf16*)w; w += NE_ * 2;   // attention output, bf16

    // fp32 -> bf16
    cvt_bf16<<<(int)(NE_ / 1024), 256, 0, stream>>>(x,  xb,  (int)NE_);
    cvt_bf16<<<(int)(EEe / 1024), 256, 0, stream>>>(Wq, wqb, (int)EEe);
    cvt_bf16<<<(int)(EEe / 1024), 256, 0, stream>>>(Wk, wkb, (int)EEe);
    cvt_bf16<<<(int)(EEe / 1024), 256, 0, stream>>>(Wv, wvb, (int)EEe);
    cvt_bf16<<<(int)(EEe / 1024), 256, 0, stream>>>(Wo, wob, (int)EEe);

    // fused QKV projection (WMMA, 32x64 tiles per wave)
    qkv_gemm<<<dim3(NTOK / 32, 3), 128, 0, stream>>>(
        xb, wqb, wkb, wvb, bq, bk, bv, qb, kb_, vb_);

    // neighborhood attention (TDM halo staging on interior tiles)
    natten<<<dim3(64, NB * NH_), 256, 0, stream>>>(qb, kb_, vb_, ab);

    // output projection (WMMA, fp32 out)
    out_gemm<<<NTOK / 32, 128, 0, stream>>>(ab, wob, bo, (float*)d_out);
}
